// HierarchicalContrastiveLoss_7928509628769
// MI455X (gfx1250) — matrix operand order repaired and checked by reading, not compiled, not run on hardware
//
#include <hip/hip_runtime.h>
#include <hip/hip_bf16.h>
#include <math.h>

typedef __attribute__((ext_vector_type(16))) _Float16 v16h;
typedef __attribute__((ext_vector_type(8)))  float    v8f;
typedef __attribute__((ext_vector_type(4)))  _Float16 v4h;

union Frag { v16h h; uint4 q[2]; };
union Acc  { v8f v; float f[8]; };

#define N_ROWS 8192
#define DIMK   128
#define TEMP_INV (1.0f / 0.07f)

// ---------------------------------------------------------------------------
// Kernel 1: row-normalize features (f32) and split each value into f16 hi + lo
// One wave (32 lanes) per row; each lane handles 4 contiguous elements.
// ---------------------------------------------------------------------------
__global__ __launch_bounds__(256) void normalize_split_kernel(
    const float* __restrict__ f, _Float16* __restrict__ fh, _Float16* __restrict__ fl) {
  const int row  = blockIdx.x * 8 + (threadIdx.x >> 5);
  const int lane = threadIdx.x & 31;
  const float4 x = *(const float4*)(f + (size_t)row * DIMK + lane * 4);
  float ss = x.x * x.x + x.y * x.y + x.z * x.z + x.w * x.w;
  #pragma unroll
  for (int m = 16; m >= 1; m >>= 1) ss += __shfl_xor(ss, m, 32);
  const float inv = 1.0f / fmaxf(sqrtf(ss), 1e-12f);
  float xn[4] = { x.x * inv, x.y * inv, x.z * inv, x.w * inv };
  union { v4h h; _Float16 e[4]; } hi, lo;
  #pragma unroll
  for (int j = 0; j < 4; ++j) {
    hi.e[j] = (_Float16)xn[j];
    lo.e[j] = (_Float16)(xn[j] - (float)hi.e[j]);
  }
  const size_t idx = (size_t)row * DIMK + lane * 4;
  *(v4h*)(fh + idx) = hi.h;
  *(v4h*)(fl + idx) = lo.h;
}

// ---------------------------------------------------------------------------
// Kernel 2: pack rel[64][64][64] (f32 0/1) into relbits[a][l] 64-bit masks
// ---------------------------------------------------------------------------
__global__ void pack_rel_kernel(const float* __restrict__ rel,
                                unsigned long long* __restrict__ relbits) {
  const int t = blockIdx.x * blockDim.x + threadIdx.x;  // 0..4095 -> (a,l)
  const float* p = rel + (size_t)t * 64;
  unsigned long long m = 0;
  for (int o = 0; o < 64; ++o)
    if (p[o] != 0.0f) m |= (1ull << o);
  relbits[t] = m;
}

// ---------------------------------------------------------------------------
// Kernel 3: posbits[b][l] = OR_a (target[b,a] ? relbits[a][l] : 0); pos_cnt
// ---------------------------------------------------------------------------
__global__ void pos_mask_kernel(const int* __restrict__ target,
                                const unsigned long long* __restrict__ relbits,
                                unsigned long long* __restrict__ posbits,
                                unsigned long long* __restrict__ pcnt) {
  const int t = blockIdx.x * blockDim.x + threadIdx.x;  // 0..8191 -> (b,l)
  const int b = t >> 6, l = t & 63;
  unsigned long long m = 0;
  for (int a = 0; a < 64; ++a)
    if (target[b * 64 + a]) m |= relbits[a * 64 + l];
  posbits[t] = m;
  atomicAdd(pcnt, (unsigned long long)__popcll(m));
}

// ---------------------------------------------------------------------------
// Kernel 4: fused GEMM (fn . fn^T via split-f16 WMMA) + masked loss epilogue.
// Grid 64x64 of 128x128 tiles, 256 threads = 8 waves, 8 16x16 accum tiles/wave.
// ---------------------------------------------------------------------------
__device__ __forceinline__ v8f wmma_f16(v16h a, v16h b, v8f c) {
  return __builtin_amdgcn_wmma_f32_16x16x32_f16(false, a, false, b, (short)0, c,
                                                false, false);
}

__global__ __launch_bounds__(256) void gemm_loss_kernel(
    const _Float16* __restrict__ fh, const _Float16* __restrict__ fl,
    const unsigned long long* __restrict__ posbits,
    double* __restrict__ pos_sum, double* __restrict__ neg_sum) {
  const int r    = blockIdx.y;
  const int c    = blockIdx.x;
  const int tid  = threadIdx.x;
  const int wave = tid >> 5;
  const int lane = tid & 31;
  const int hf   = lane >> 4;    // lane half (0: lanes 0-15, 1: lanes 16-31)
  const int ln   = lane & 15;

  // A fragment row for this lane (M = lane & 15 within the wave's 16-row strip)
  const int rowA = r * 128 + wave * 16 + ln;

  v8f acc[8] = {};

  #pragma unroll
  for (int kb = 0; kb < DIMK; kb += 32) {
    // A fragment (16-bit A 16x32): u[0..3] = K kb+8h..+7, u[4..7] = K kb+16+8h..+7
    Frag ahi, alo;
    {
      const _Float16* pa = fh + (size_t)rowA * DIMK + kb + hf * 8;
      ahi.q[0] = *(const uint4*)(pa);
      ahi.q[1] = *(const uint4*)(pa + 16);
      const _Float16* pal = fl + (size_t)rowA * DIMK + kb + hf * 8;
      alo.q[0] = *(const uint4*)(pal);
      alo.q[1] = *(const uint4*)(pal + 16);
    }
    #pragma unroll
    for (int t = 0; t < 8; ++t) {
      // B fragment (16-bit B 32x16): lane holds column N=ln, K kb+8h..+7 then kb+16+8h..+7
      const int rowB = c * 128 + t * 16 + ln;
      Frag bhi, blo;
      const _Float16* pb = fh + (size_t)rowB * DIMK + kb + hf * 8;
      bhi.q[0] = *(const uint4*)(pb);
      bhi.q[1] = *(const uint4*)(pb + 16);
      const _Float16* pbl = fl + (size_t)rowB * DIMK + kb + hf * 8;
      blo.q[0] = *(const uint4*)(pbl);
      blo.q[1] = *(const uint4*)(pbl + 16);
      // dot(a,b) ~= hi.hi + hi.lo + lo.hi  (lo.lo negligible)
      acc[t] = wmma_f16(ahi.h, bhi.h, acc[t]);
      acc[t] = wmma_f16(ahi.h, blo.h, acc[t]);
      acc[t] = wmma_f16(alo.h, bhi.h, acc[t]);
    }
  }

  // Epilogue: dist = 1 - dot; positives only on diagonal blocks & same sample.
  const bool diag = (r == c);
  float psum = 0.0f, nsum = 0.0f;
  int ilocs[8];
  unsigned long long pm[8];
  #pragma unroll
  for (int v = 0; v < 8; ++v) {
    const int M = v + (hf ? 8 : 0);           // C/D layout: M = vgpr + 8*(lane>=16)
    const int iloc = wave * 16 + M;           // row within 128-tile
    ilocs[v] = iloc;
    pm[v] = diag ? posbits[(size_t)(r * 2 + (iloc >> 6)) * 64 + (iloc & 63)] : 0ull;
  }
  #pragma unroll
  for (int t = 0; t < 8; ++t) {
    Acc u; u.v = acc[t];
    const int sj = t >> 2;                    // sample half of the column
    const int o  = (t & 3) * 16 + ln;         // j within its 64-sample block
    #pragma unroll
    for (int v = 0; v < 8; ++v) {
      const float dist = 1.0f - u.f[v];
      const bool ispos = diag && ((ilocs[v] >> 6) == sj) && ((pm[v] >> o) & 1ull);
      if (ispos) { psum += dist; nsum += 1.0f; }   // exp(0) = 1 for positives
      else       { nsum += __expf(dist * (-TEMP_INV)); }
    }
  }

  // Block reduction, then double atomics (sum magnitude ~6.7e7 needs f64).
  __shared__ float sp[256], sn[256];
  sp[tid] = psum; sn[tid] = nsum;
  __syncthreads();
  for (int s = 128; s > 0; s >>= 1) {
    if (tid < s) { sp[tid] += sp[tid + s]; sn[tid] += sn[tid + s]; }
    __syncthreads();
  }
  if (tid == 0) {
    atomicAdd(pos_sum, (double)sp[0]);
    atomicAdd(neg_sum, (double)sn[0]);
  }
}

// ---------------------------------------------------------------------------
// Kernel 5: finalize scalar loss
// ---------------------------------------------------------------------------
__global__ void finalize_kernel(const double* __restrict__ pos_sum,
                                const double* __restrict__ neg_sum,
                                const unsigned long long* __restrict__ pcnt,
                                float* __restrict__ out) {
  const double pc = (double)*pcnt;
  const double nn = (double)N_ROWS * (double)N_ROWS;
  const double pos_loss = *pos_sum / pc;
  const double neg_mean = *neg_sum / (nn - pc);
  const double lg = log(neg_mean + 1e-6);
  out[0] = (float)(pos_loss + (lg > 0.0 ? lg : 0.0));
}

// ---------------------------------------------------------------------------
extern "C" void kernel_launch(void* const* d_in, const int* in_sizes, int n_in,
                              void* d_out, int out_size, void* d_ws, size_t ws_size,
                              hipStream_t stream) {
  const float* features = (const float*)d_in[0];   // [128,64,128] f32
  const float* rel      = (const float*)d_in[1];   // [64,64,64]   f32
  const int*   target   = (const int*)d_in[2];     // [128,64]     i32
  float* out = (float*)d_out;

  char* ws = (char*)d_ws;
  double*             pos_sum = (double*)(ws + 0);
  double*             neg_sum = (double*)(ws + 8);
  unsigned long long* pcnt    = (unsigned long long*)(ws + 16);
  unsigned long long* relbits = (unsigned long long*)(ws + 64);                 // 32 KB
  unsigned long long* posbits = (unsigned long long*)(ws + 64 + 32768);         // 64 KB
  _Float16*           fh      = (_Float16*)(ws + 64 + 32768 + 65536);           // 2 MB
  _Float16*           fl      = (_Float16*)(ws + 64 + 32768 + 65536 + 2097152); // 2 MB

  hipMemsetAsync(ws, 0, 64, stream);  // zero accumulators (graph-capturable)

  normalize_split_kernel<<<N_ROWS / 8, 256, 0, stream>>>(features, fh, fl);
  pack_rel_kernel<<<16, 256, 0, stream>>>(rel, relbits);
  pos_mask_kernel<<<32, 256, 0, stream>>>(target, relbits, posbits, pcnt);

  dim3 grid(64, 64);
  gemm_loss_kernel<<<grid, 256, 0, stream>>>(fh, fl, posbits, pos_sum, neg_sum);

  finalize_kernel<<<1, 1, 0, stream>>>(pos_sum, neg_sum, pcnt, out);
}
